// SpectralCell_10986526343850
// MI455X (gfx1250) — matrix operand order repaired and checked by reading, not compiled, not run on hardware
//
#include <hip/hip_runtime.h>
#include <math.h>

typedef __attribute__((ext_vector_type(16))) _Float16 v16h;
typedef __attribute__((ext_vector_type(8)))  float    v8f;

#define TOKEN_DIM 64
#define HIDDEN    128
#define VR        16
#define DM        8
#define MAT_DIM   128
#define NTOK      (64 * 512)
#define WV        4   // waves per MLP block

// ---------------- common device helpers ----------------

// Branch-free erf (Abramowitz-Stegun 7.1.26, |err| < 1.5e-7): straight-line
// FMA chain + v_rcp_f32 + v_exp_f32, no EXEC save/restore, no IEEE div expansion.
__device__ __forceinline__ float erf_fast(float x) {
    float ax = fabsf(x);
    float t  = __builtin_amdgcn_rcpf(1.0f + 0.3275911f * ax);
    float y  = t * (0.254829592f +
               t * (-0.284496736f +
               t * (1.421413741f +
               t * (-1.453152027f +
               t * 1.061405429f))));
    float r = 1.0f - y * __expf(-ax * ax);
    return copysignf(r, x);
}

__device__ __forceinline__ float gelu_f(float x) {
    return 0.5f * x * (1.0f + erf_fast(x * 0.7071067811865476f));
}

__device__ __forceinline__ v8f wmma16(v16h a, v16h b, v8f c) {
    return __builtin_amdgcn_wmma_f32_16x16x32_f16(false, a, false, b, (short)0, c, false, false);
}

// A fragment (16x32 f16) from per-wave LDS activations, row-major [16][ldk] f32.
// ISA layout: lanes 0-15 hold M=lane, K = ks*32 + {0..7, 16..23}; lanes 16-31 +8.
__device__ __forceinline__ v16h frag_a_lds(const float* act, int ldk, int kstep, int lane) {
    int m  = lane & 15;
    int kb = kstep * 32 + ((lane >> 4) << 3);
    const float* r = act + m * ldk + kb;
    v16h a;
#pragma unroll
    for (int e = 0; e < 8; ++e) a[e] = (_Float16)r[e];
#pragma unroll
    for (int e = 0; e < 8; ++e) a[8 + e] = (_Float16)r[16 + e];
    return a;
}

// A fragment from f16 LDS staging (pure intermediates): two ds_load_b128, no cvt.
__device__ __forceinline__ v16h frag_a_lds16(const _Float16* act, int ldk, int kstep, int lane) {
    int m  = lane & 15;
    int kb = kstep * 32 + ((lane >> 4) << 3);
    const _Float16* r = act + m * ldk + kb;
    v16h a;
#pragma unroll
    for (int e = 0; e < 8; ++e) a[e] = r[e];
#pragma unroll
    for (int e = 0; e < 8; ++e) a[8 + e] = r[16 + e];
    return a;
}

// Same as frag_a_lds but applies LayerNorm on the fly.
__device__ __forceinline__ v16h frag_a_lds_ln(const float* act, int ldk, int kstep, int lane,
                                              float mu, float rs,
                                              const float* __restrict__ g,
                                              const float* __restrict__ b) {
    int m  = lane & 15;
    int kb = kstep * 32 + ((lane >> 4) << 3);
    const float* r = act + m * ldk + kb;
    v16h a;
#pragma unroll
    for (int e = 0; e < 8; ++e)
        a[e] = (_Float16)((r[e] - mu) * rs * g[kb + e] + b[kb + e]);
#pragma unroll
    for (int e = 0; e < 8; ++e)
        a[8 + e] = (_Float16)((r[16 + e] - mu) * rs * g[kb + 16 + e] + b[kb + 16 + e]);
    return a;
}

// B fragment (32x16 f16) from pre-converted row-major f16 weights W[Nout][Kin]
// (y = x @ W^T). Each lane reads K = {0..7, 16..23} (+8 for lanes 16-31) -> 2x global_load_b128.
__device__ __forceinline__ v16h frag_b_f16(const _Float16* __restrict__ W, int Kin,
                                           int ntile, int kstep, int lane) {
    int n  = ntile * 16 + (lane & 15);
    int k0 = kstep * 32 + ((lane >> 4) << 3);
    const _Float16* p = W + n * Kin + k0;
    v16h b;
#pragma unroll
    for (int e = 0; e < 8; ++e) b[e] = p[e];
#pragma unroll
    for (int e = 0; e < 8; ++e) b[8 + e] = p[16 + e];
    return b;
}

// C accumulator init with bias (bias depends only on output column n).
__device__ __forceinline__ v8f frag_c_bias(const float* __restrict__ bias, int ntile, int lane) {
    float v = bias[ntile * 16 + (lane & 15)];
    v8f c;
#pragma unroll
    for (int r = 0; r < 8; ++r) c[r] = v;
    return c;
}

// C/D layout: col n = lane&15, rows m = r + 8*(lane>=16).
__device__ __forceinline__ void store_c_lds(float* act, int ldk, int ntile, int lane,
                                            v8f c, bool do_gelu) {
    int n  = ntile * 16 + (lane & 15);
    int mb = (lane >> 4) << 3;
#pragma unroll
    for (int r = 0; r < 8; ++r) {
        float v = c[r];
        if (do_gelu) v = gelu_f(v);
        act[(mb + r) * ldk + n] = v;
    }
}

// GELU + store as f16 (for tiles consumed only as future A fragments).
__device__ __forceinline__ void store_c_lds16_gelu(_Float16* act, int ldk, int ntile, int lane,
                                                   v8f c) {
    int n  = ntile * 16 + (lane & 15);
    int mb = (lane >> 4) << 3;
#pragma unroll
    for (int r = 0; r < 8; ++r) act[(mb + r) * ldk + n] = (_Float16)gelu_f(c[r]);
}

__device__ __forceinline__ void add_c_lds(float* act, int ldk, int ntile, int lane, v8f c) {
    int n  = ntile * 16 + (lane & 15);
    int mb = (lane >> 4) << 3;
#pragma unroll
    for (int r = 0; r < 8; ++r) act[(mb + r) * ldk + n] += c[r];
}

// Linear layer, output -> f32 LDS (optionally GELU)
template <int KSTEPS, int NTILES>
__device__ __forceinline__ void lin_layer_f32(const v16h* afr, const _Float16* __restrict__ W,
                                              const float* __restrict__ bias, int Kin,
                                              float* out_act, int ldk, int lane, bool do_gelu) {
#pragma unroll
    for (int nt = 0; nt < NTILES; ++nt) {
        v8f acc = frag_c_bias(bias, nt, lane);
#pragma unroll
        for (int ks = 0; ks < KSTEPS; ++ks)
            acc = wmma16(afr[ks], frag_b_f16(W, Kin, nt, ks, lane), acc);
        store_c_lds(out_act, ldk, nt, lane, acc, do_gelu);
    }
}

// Linear layer, GELU, output -> f16 LDS staging
template <int KSTEPS, int NTILES>
__device__ __forceinline__ void lin_layer_f16(const v16h* afr, const _Float16* __restrict__ W,
                                              const float* __restrict__ bias, int Kin,
                                              _Float16* out_act, int ldk, int lane) {
#pragma unroll
    for (int nt = 0; nt < NTILES; ++nt) {
        v8f acc = frag_c_bias(bias, nt, lane);
#pragma unroll
        for (int ks = 0; ks < KSTEPS; ++ks)
            acc = wmma16(afr[ks], frag_b_f16(W, Kin, nt, ks, lane), acc);
        store_c_lds16_gelu(out_act, ldk, nt, lane, acc);
    }
}

// Per-token mean / inv-std over HIDDEN features; lane pair (L, L^16) shares token L&15.
__device__ __forceinline__ void token_stats(const float* act, int ldk, int lane,
                                            float& mu, float& rs) {
    int t    = lane & 15;
    int half = lane >> 4;
    const float* r = act + t * ldk + half * 64;
    float s = 0.f;
    for (int i = 0; i < 64; ++i) s += r[i];
    s += __shfl_xor(s, 16, 32);
    float m = s * (1.0f / 128.0f);
    float v = 0.f;
    for (int i = 0; i < 64; ++i) { float d = r[i] - m; v += d * d; }
    v += __shfl_xor(v, 16, 32);
    mu = m;
    rs = rsqrtf(v * (1.0f / 128.0f) + 1e-5f);
}

__device__ __forceinline__ void res_block(float* hb, _Float16* t16, int lane,
                                          const float* lng, const float* lnb,
                                          const _Float16* w1, const float* b1,
                                          const _Float16* w2, const float* b2) {
    float mu, rs;
    token_stats(hb, HIDDEN, lane, mu, rs);
    v16h a[4];
#pragma unroll
    for (int ks = 0; ks < 4; ++ks) a[ks] = frag_a_lds_ln(hb, HIDDEN, ks, lane, mu, rs, lng, lnb);
    lin_layer_f16<4, 8>(a, w1, b1, HIDDEN, t16, HIDDEN, lane);
    __syncthreads();
    v16h a2[4];
#pragma unroll
    for (int ks = 0; ks < 4; ++ks) a2[ks] = frag_a_lds16(t16, HIDDEN, ks, lane);
#pragma unroll
    for (int nt = 0; nt < 8; ++nt) {
        v8f acc = frag_c_bias(b2, nt, lane);
#pragma unroll
        for (int ks = 0; ks < 4; ++ks)
            acc = wmma16(a2[ks], frag_b_f16(w2, HIDDEN, nt, ks, lane), acc);
        add_c_lds(hb, HIDDEN, nt, lane, acc);
    }
    __syncthreads();
}

// ---------------- kernel 0: one-time weight f32 -> f16 conversion ----------------

struct WcvtP { const float* src[12]; _Float16* dst; };

// f16 weight pool offsets (elements)
#define WO_ENC_IN   0
#define WO_EB0_W1   8192
#define WO_EB0_W2   24576
#define WO_EB1_W1   40960
#define WO_EB1_W2   57344
#define WO_ENC_OUT  73728
#define WO_OB0_W1   90112
#define WO_OB0_W2   106496
#define WO_OB1_W1   122880
#define WO_OB1_W2   139264
#define WO_OUT_IN   155648
#define WO_OUT_PROJ 172032

__global__ __launch_bounds__(256)
void sc_wcvt_kernel(WcvtP P) {
    const int offs[12]  = {WO_ENC_IN, WO_EB0_W1, WO_EB0_W2, WO_EB1_W1, WO_EB1_W2, WO_ENC_OUT,
                           WO_OB0_W1, WO_OB0_W2, WO_OB1_W1, WO_OB1_W2, WO_OUT_IN, WO_OUT_PROJ};
    const int sizes[12] = {8192, 16384, 16384, 16384, 16384, 16384,
                           16384, 16384, 16384, 16384, 16384, 8192};
    int m = blockIdx.x;
    const float* s = P.src[m];
    _Float16*    d = P.dst + offs[m];
    int n = sizes[m];
    for (int i = threadIdx.x; i < n; i += 256) d[i] = (_Float16)s[i];
}

// ---------------- kernel 1: encoder MLP -> normalized M ----------------

struct EncP {
    const _Float16 *enc_in_w; const float *enc_in_b;
    const float *b0_lng, *b0_lnb; const _Float16 *b0_w1; const float *b0_b1;
    const _Float16 *b0_w2; const float *b0_b2;
    const float *b1_lng, *b1_lnb; const _Float16 *b1_w1; const float *b1_b1;
    const _Float16 *b1_w2; const float *b1_b2;
    const _Float16 *enc_out_w; const float *enc_out_b;
};

__global__ __launch_bounds__(WV * 32)
void sc_enc_kernel(const float* __restrict__ tokens, float* __restrict__ Mout, EncP P) {
    __shared__ float    hb_s[WV][16][HIDDEN];   // residual stream (f32)
    __shared__ _Float16 t16_s[WV][16][HIDDEN];  // f16 staging for A-only intermediates
    int lane = threadIdx.x & 31;
    int w    = threadIdx.x >> 5;
    float*    hb  = &hb_s[w][0][0];
    _Float16* t16 = &t16_s[w][0][0];
    int tok0 = (blockIdx.x * WV + w) * 16;

    // stage tokens as f16 (they are consumed only as A fragments)
    for (int i = lane; i < 16 * TOKEN_DIM; i += 32) {
        int t = i >> 6, k = i & 63;
        t16[t * HIDDEN + k] = (_Float16)tokens[(size_t)(tok0 + t) * TOKEN_DIM + k];
    }
    __syncthreads();

    // enc_in: 64 -> 128, exact GELU, into f32 residual stream
    v16h a[4];
    a[0] = frag_a_lds16(t16, HIDDEN, 0, lane);
    a[1] = frag_a_lds16(t16, HIDDEN, 1, lane);
    lin_layer_f32<2, 8>(a, P.enc_in_w, P.enc_in_b, TOKEN_DIM, hb, HIDDEN, lane, true);
    __syncthreads();

    res_block(hb, t16, lane, P.b0_lng, P.b0_lnb, P.b0_w1, P.b0_b1, P.b0_w2, P.b0_b2);
    res_block(hb, t16, lane, P.b1_lng, P.b1_lnb, P.b1_w1, P.b1_b1, P.b1_w2, P.b1_b2);

    // enc_out: 128 -> 128 (no activation); A frags are in registers, so we can
    // overwrite hb with the output tiles.
#pragma unroll
    for (int ks = 0; ks < 4; ++ks) a[ks] = frag_a_lds(hb, HIDDEN, ks, lane);
    lin_layer_f32<4, 8>(a, P.enc_out_w, P.enc_out_b, HIDDEN, hb, HIDDEN, lane, false);
    __syncthreads();

    // normalize each 8-dim vertex row, write M
    int t  = lane & 15;
    int rh = lane >> 4;
#pragma unroll
    for (int j = 0; j < 8; ++j) {
        int v = rh * 8 + j;
        const float* row = hb + t * HIDDEN + v * DM;
        float s = 0.f;
#pragma unroll
        for (int e = 0; e < DM; ++e) s += row[e] * row[e];
        float inv = 1.0f / fmaxf(sqrtf(s), 1e-12f);
        size_t base = (size_t)(tok0 + t) * MAT_DIM + v * DM;
#pragma unroll
        for (int e = 0; e < DM; ++e) Mout[base + e] = row[e] * inv;
    }
}

// ---------------- kernel 2: Cayley-Menger + Gram Jacobi SVD ----------------

__global__ __launch_bounds__(256)
void sc_svd_kernel(const float* __restrict__ Mglob, float* __restrict__ Uout,
                   float* __restrict__ Vtout, float* __restrict__ Sout,
                   float* __restrict__ d2out, float* __restrict__ vol2out) {
    int tok = blockIdx.x * blockDim.x + threadIdx.x;
    if (tok >= NTOK) return;
    const float* M = Mglob + (size_t)tok * MAT_DIM;

    // --- Cayley-Menger on vertices {0,3,7,11,15} ---
    const int idx[5] = {0, 3, 7, 11, 15};
    float d2[5][5];
    for (int i = 0; i < 5; ++i)
        for (int j = 0; j < 5; ++j) {
            float gij = 0, gii = 0, gjj = 0;
            for (int e = 0; e < DM; ++e) {
                float vi = M[idx[i] * DM + e], vj = M[idx[j] * DM + e];
                gij += vi * vj; gii += vi * vi; gjj += vj * vj;
            }
            float v = gii + gjj - 2.0f * gij;
            d2[i][j] = v > 0.f ? v : 0.f;
        }
    int pc = 0;
    for (int i = 0; i < 5; ++i)
        for (int j = i + 1; j < 5; ++j)
            d2out[(size_t)tok * 10 + pc++] = d2[i][j];

    float cm[6][6];
    cm[0][0] = 0.f;
    for (int i = 1; i < 6; ++i) { cm[0][i] = 1.f; cm[i][0] = 1.f; }
    for (int i = 0; i < 5; ++i)
        for (int j = 0; j < 5; ++j) cm[i + 1][j + 1] = d2[i][j];
    float det = 1.f;
    for (int c = 0; c < 6; ++c) {
        int piv = c; float mx = fabsf(cm[c][c]);
        for (int r2 = c + 1; r2 < 6; ++r2) {
            float aa = fabsf(cm[r2][c]);
            if (aa > mx) { mx = aa; piv = r2; }
        }
        if (piv != c) {
            for (int k = 0; k < 6; ++k) { float t = cm[c][k]; cm[c][k] = cm[piv][k]; cm[piv][k] = t; }
            det = -det;
        }
        float p = cm[c][c];
        det *= p;
        if (p != 0.f) {
            float ip = 1.0f / p;
            for (int r2 = c + 1; r2 < 6; ++r2) {
                float f = cm[r2][c] * ip;
                for (int k = c; k < 6; ++k) cm[r2][k] -= f * cm[c][k];
            }
        }
    }
    vol2out[tok] = (-1.0f / 9216.0f) * det;  // PREFACTOR, CM_K=4

    // --- Gram + cyclic Jacobi eigensolver ---
    float G[8][8];
    for (int i = 0; i < 8; ++i)
        for (int j = 0; j < 8; ++j) G[i][j] = 0.f;
    for (int v = 0; v < VR; ++v) {
        float row[8];
        for (int e = 0; e < 8; ++e) row[e] = M[v * DM + e];
        for (int i = 0; i < 8; ++i)
            for (int j = 0; j < 8; ++j) G[i][j] += row[i] * row[j];
    }
    for (int i = 0; i < 8; ++i) G[i][i] += 1e-12f;
    float Vv[8][8];
    for (int i = 0; i < 8; ++i)
        for (int j = 0; j < 8; ++j) Vv[i][j] = (i == j) ? 1.f : 0.f;
    for (int sweep = 0; sweep < 12; ++sweep) {
        for (int p = 0; p < 7; ++p)
            for (int q = p + 1; q < 8; ++q) {
                float apq = G[p][q];
                if (fabsf(apq) < 1e-30f) continue;
                float tau = (G[q][q] - G[p][p]) / (2.f * apq);
                float tt  = (tau >= 0.f) ? 1.f / (tau + sqrtf(1.f + tau * tau))
                                         : 1.f / (tau - sqrtf(1.f + tau * tau));
                float cth = rsqrtf(1.f + tt * tt);
                float sth = tt * cth;
                for (int k = 0; k < 8; ++k) {
                    float gkp = G[k][p], gkq = G[k][q];
                    G[k][p] = cth * gkp - sth * gkq;
                    G[k][q] = sth * gkp + cth * gkq;
                }
                for (int k = 0; k < 8; ++k) {
                    float gpk = G[p][k], gqk = G[q][k];
                    G[p][k] = cth * gpk - sth * gqk;
                    G[q][k] = sth * gpk + cth * gqk;
                }
                for (int k = 0; k < 8; ++k) {
                    float vkp = Vv[k][p], vkq = Vv[k][q];
                    Vv[k][p] = cth * vkp - sth * vkq;
                    Vv[k][q] = sth * vkp + cth * vkq;
                }
            }
    }
    float wv[8]; int ord[8];
    for (int i = 0; i < 8; ++i) { wv[i] = G[i][i]; ord[i] = i; }
    for (int i = 0; i < 8; ++i)
        for (int j = i + 1; j < 8; ++j)
            if (wv[ord[j]] > wv[ord[i]]) { int t = ord[i]; ord[i] = ord[j]; ord[j] = t; }
    float S[8];
    for (int d = 0; d < 8; ++d) {
        float wd = wv[ord[d]];
        S[d] = sqrtf(wd > 1e-24f ? wd : 1e-24f);
        Sout[(size_t)tok * DM + d] = S[d];
    }
    for (int d = 0; d < 8; ++d) {
        int od = ord[d];
        float inv = 1.0f / (S[d] > 1e-16f ? S[d] : 1e-16f);
        for (int e = 0; e < 8; ++e) Vtout[(size_t)tok * 64 + d * 8 + e] = Vv[e][od];
        for (int v = 0; v < VR; ++v) {
            float acc = 0.f;
            for (int e = 0; e < 8; ++e) acc += M[v * DM + e] * Vv[e][od];
            Uout[(size_t)tok * MAT_DIM + v * DM + d] = acc * inv;
        }
    }
}

// ---------------- kernel 3: cross attention gate on singular values ----------------

__global__ __launch_bounds__(512)
void sc_attn_kernel(const float* __restrict__ Sin, float* __restrict__ Scout,
                    const float* __restrict__ lng, const float* __restrict__ lnb,
                    const float* __restrict__ wqkv, const float* __restrict__ bqkv,
                    const float* __restrict__ wout, const float* __restrict__ bout,
                    const float* __restrict__ alphap) {
    __shared__ float Ks[2][512][4];
    __shared__ float Vs[2][512][4];
    int n = threadIdx.x;
    int b = blockIdx.x;
    const float* Srow = Sin + ((size_t)b * 512 + n) * DM;
    float s[8];
#pragma unroll
    for (int d = 0; d < 8; ++d) s[d] = Srow[d];
    float mu = 0.f;
#pragma unroll
    for (int d = 0; d < 8; ++d) mu += s[d];
    mu *= 0.125f;
    float var = 0.f;
#pragma unroll
    for (int d = 0; d < 8; ++d) { float dd = s[d] - mu; var += dd * dd; }
    float rs = rsqrtf(var * 0.125f + 1e-5f);
    float sn[8];
#pragma unroll
    for (int d = 0; d < 8; ++d) sn[d] = (s[d] - mu) * rs * lng[d] + lnb[d];

    float q[8];
#pragma unroll
    for (int o = 0; o < 24; ++o) {
        float acc = bqkv[o];
#pragma unroll
        for (int i = 0; i < 8; ++i) acc += sn[i] * wqkv[o * 8 + i];
        if (o < 8)       q[o] = acc;
        else if (o < 16) Ks[(o - 8) >> 2][n][(o - 8) & 3] = acc;
        else             Vs[(o - 16) >> 2][n][(o - 16) & 3] = acc;
    }
    __syncthreads();

    float outv[8];
#pragma unroll
    for (int h = 0; h < 2; ++h) {
        float q0 = q[h * 4 + 0] * 0.5f, q1 = q[h * 4 + 1] * 0.5f;
        float q2 = q[h * 4 + 2] * 0.5f, q3 = q[h * 4 + 3] * 0.5f;
        float mx = -1e30f;
        for (int m = 0; m < 512; ++m) {
            float l = q0 * Ks[h][m][0] + q1 * Ks[h][m][1] + q2 * Ks[h][m][2] + q3 * Ks[h][m][3];
            mx = fmaxf(mx, l);
        }
        float sum = 0.f, o0 = 0.f, o1 = 0.f, o2 = 0.f, o3 = 0.f;
        for (int m = 0; m < 512; ++m) {
            float l = q0 * Ks[h][m][0] + q1 * Ks[h][m][1] + q2 * Ks[h][m][2] + q3 * Ks[h][m][3];
            float p = __expf(l - mx);
            sum += p;
            o0 += p * Vs[h][m][0]; o1 += p * Vs[h][m][1];
            o2 += p * Vs[h][m][2]; o3 += p * Vs[h][m][3];
        }
        float inv = __builtin_amdgcn_rcpf(sum);
        outv[h * 4 + 0] = o0 * inv; outv[h * 4 + 1] = o1 * inv;
        outv[h * 4 + 2] = o2 * inv; outv[h * 4 + 3] = o3 * inv;
    }

    float* Scrow = Scout + ((size_t)b * 512 + n) * DM;
#pragma unroll
    for (int d = 0; d < 8; ++d) {
        float g = bout[d];
#pragma unroll
        for (int i = 0; i < 8; ++i) g += outv[i] * wout[d * 8 + i];
        g = tanhf(g);
        float al = 0.2f / (1.0f + __expf(-alphap[d]));
        Scrow[d] = s[d] * (1.0f + al * g);
    }
}

// ---------------- kernel 4: reconstruct M_hat + decoder MLP ----------------

struct DecP {
    const _Float16 *out_in_w; const float *out_in_b;
    const float *b0_lng, *b0_lnb; const _Float16 *b0_w1; const float *b0_b1;
    const _Float16 *b0_w2; const float *b0_b2;
    const float *b1_lng, *b1_lnb; const _Float16 *b1_w1; const float *b1_b1;
    const _Float16 *b1_w2; const float *b1_b2;
    const _Float16 *out_proj_w; const float *out_proj_b;
};

__global__ __launch_bounds__(WV * 32)
void sc_dec_kernel(const float* __restrict__ U, const float* __restrict__ Vt,
                   const float* __restrict__ Sc, float* __restrict__ Out, DecP P) {
    __shared__ float    hb_s[WV][16][HIDDEN];
    __shared__ _Float16 t16_s[WV][16][HIDDEN];
    int lane = threadIdx.x & 31;
    int w    = threadIdx.x >> 5;
    float*    hb  = &hb_s[w][0][0];
    _Float16* t16 = &t16_s[w][0][0];
    int tok0 = (blockIdx.x * WV + w) * 16;

    // M_hat = U * diag(Sc) * Vt ; lane pair per token, 8 vertex rows each.
    // M_hat is consumed only as A fragments -> stage directly as f16.
    {
        int t   = lane & 15;
        int rh  = lane >> 4;
        int tok = tok0 + t;
        float sv[8], vt[8][8];
#pragma unroll
        for (int d = 0; d < 8; ++d) sv[d] = Sc[(size_t)tok * DM + d];
#pragma unroll
        for (int d = 0; d < 8; ++d)
#pragma unroll
            for (int e = 0; e < 8; ++e) vt[d][e] = Vt[(size_t)tok * 64 + d * 8 + e];
#pragma unroll
        for (int j = 0; j < 8; ++j) {
            int v = rh * 8 + j;
            float u[8];
#pragma unroll
            for (int d = 0; d < 8; ++d) u[d] = U[(size_t)tok * MAT_DIM + v * DM + d] * sv[d];
#pragma unroll
            for (int e = 0; e < 8; ++e) {
                float acc = 0.f;
#pragma unroll
                for (int d = 0; d < 8; ++d) acc += u[d] * vt[d][e];
                t16[t * HIDDEN + v * DM + e] = (_Float16)acc;
            }
        }
    }
    __syncthreads();

    v16h a[4];
#pragma unroll
    for (int ks = 0; ks < 4; ++ks) a[ks] = frag_a_lds16(t16, HIDDEN, ks, lane);
    lin_layer_f32<4, 8>(a, P.out_in_w, P.out_in_b, HIDDEN, hb, HIDDEN, lane, true);
    __syncthreads();

    res_block(hb, t16, lane, P.b0_lng, P.b0_lnb, P.b0_w1, P.b0_b1, P.b0_w2, P.b0_b2);
    res_block(hb, t16, lane, P.b1_lng, P.b1_lnb, P.b1_w1, P.b1_b1, P.b1_w2, P.b1_b2);

    // out_proj: 128 -> 64, write straight to global
#pragma unroll
    for (int ks = 0; ks < 4; ++ks) a[ks] = frag_a_lds(hb, HIDDEN, ks, lane);
#pragma unroll
    for (int nt = 0; nt < 4; ++nt) {
        v8f acc = frag_c_bias(P.out_proj_b, nt, lane);
#pragma unroll
        for (int ks = 0; ks < 4; ++ks)
            acc = wmma16(a[ks], frag_b_f16(P.out_proj_w, HIDDEN, nt, ks, lane), acc);
        int n  = nt * 16 + (lane & 15);
        int mb = (lane >> 4) << 3;
#pragma unroll
        for (int r = 0; r < 8; ++r)
            Out[(size_t)(tok0 + mb + r) * TOKEN_DIM + n] = acc[r];
    }
}

// ---------------- host launcher ----------------

extern "C" void kernel_launch(void* const* d_in, const int* in_sizes, int n_in,
                              void* d_out, int out_size, void* d_ws, size_t ws_size,
                              hipStream_t stream) {
    // Identify the tokens input by its unique element count; remaining inputs are
    // the params pytree leaves in JAX sorted-key flatten order.
    int tok_idx = 0;
    for (int i = 0; i < n_in; ++i)
        if (in_sizes[i] == 64 * 512 * 64) { tok_idx = i; break; }
    const float* prm[64];
    int np = 0;
    for (int i = 0; i < n_in && np < 64; ++i)
        if (i != tok_idx) prm[np++] = (const float*)d_in[i];
    const float* tokens = (const float*)d_in[tok_idx];

    // Output tuple layout (flat, in return order)
    float* out   = (float*)d_out;             // (B,N,64)
    float* Mout  = out + 2097152;             // (B,N,16,8)
    float* Sorig = out + 6291456;             // (B,N,8)
    float* Scout = out + 6553600;             // (B,N,8)
    float* d2out = out + 6815744;             // (B*N,10)
    float* v2out = out + 7143424;             // (B*N,)

    // Workspace: U (16 MB) + Vt (8 MB) + f16 weight pool (352 KB)
    float*    Uws   = (float*)d_ws;
    float*    Vtws  = Uws + 4194304;
    _Float16* wpool = (_Float16*)(Vtws + 2097152);

    // params leaf order (sorted keys):
    // 0 alpha 1 cr_ln_b 2 cr_ln_g 3 cr_out_b 4 cr_out_w 5 qkv_b 6 qkv_w
    // 7..12  enc_blocks[0]: l1.b l1.w l2.b l2.w ln_b ln_g
    // 13..18 enc_blocks[1]
    // 19 enc_in.b 20 enc_in.w 21 enc_out.b 22 enc_out.w
    // 23..28 out_blocks[0], 29..34 out_blocks[1]
    // 35 out_in.b 36 out_in.w 37 out_proj.b 38 out_proj.w
    WcvtP wp;
    wp.src[0]  = prm[20]; // enc_in.w
    wp.src[1]  = prm[8];  // eb0.l1.w
    wp.src[2]  = prm[10]; // eb0.l2.w
    wp.src[3]  = prm[14]; // eb1.l1.w
    wp.src[4]  = prm[16]; // eb1.l2.w
    wp.src[5]  = prm[22]; // enc_out.w
    wp.src[6]  = prm[24]; // ob0.l1.w
    wp.src[7]  = prm[26]; // ob0.l2.w
    wp.src[8]  = prm[30]; // ob1.l1.w
    wp.src[9]  = prm[32]; // ob1.l2.w
    wp.src[10] = prm[36]; // out_in.w
    wp.src[11] = prm[38]; // out_proj.w
    wp.dst = wpool;

    EncP ep;
    ep.enc_in_w = wpool + WO_ENC_IN;  ep.enc_in_b = prm[19];
    ep.b0_lng = prm[12]; ep.b0_lnb = prm[11];
    ep.b0_w1  = wpool + WO_EB0_W1;   ep.b0_b1 = prm[7];
    ep.b0_w2  = wpool + WO_EB0_W2;   ep.b0_b2 = prm[9];
    ep.b1_lng = prm[18]; ep.b1_lnb = prm[17];
    ep.b1_w1  = wpool + WO_EB1_W1;   ep.b1_b1 = prm[13];
    ep.b1_w2  = wpool + WO_EB1_W2;   ep.b1_b2 = prm[15];
    ep.enc_out_w = wpool + WO_ENC_OUT; ep.enc_out_b = prm[21];

    DecP dp;
    dp.out_in_w = wpool + WO_OUT_IN;  dp.out_in_b = prm[35];
    dp.b0_lng = prm[28]; dp.b0_lnb = prm[27];
    dp.b0_w1  = wpool + WO_OB0_W1;   dp.b0_b1 = prm[23];
    dp.b0_w2  = wpool + WO_OB0_W2;   dp.b0_b2 = prm[25];
    dp.b1_lng = prm[34]; dp.b1_lnb = prm[33];
    dp.b1_w1  = wpool + WO_OB1_W1;   dp.b1_b1 = prm[29];
    dp.b1_w2  = wpool + WO_OB1_W2;   dp.b1_b2 = prm[31];
    dp.out_proj_w = wpool + WO_OUT_PROJ; dp.out_proj_b = prm[37];

    dim3 mlp_grid(NTOK / (WV * 16));
    dim3 mlp_block(WV * 32);

    sc_wcvt_kernel<<<12, 256, 0, stream>>>(wp);
    sc_enc_kernel<<<mlp_grid, mlp_block, 0, stream>>>(tokens, Mout, ep);
    sc_svd_kernel<<<NTOK / 256, 256, 0, stream>>>(Mout, Uws, Vtws, Sorig, d2out, v2out);
    sc_attn_kernel<<<64, 512, 0, stream>>>(Sorig, Scout,
                                           prm[2], prm[1], prm[6], prm[5],
                                           prm[4], prm[3], prm[0]);
    sc_dec_kernel<<<mlp_grid, mlp_block, 0, stream>>>(Uws, Vtws, Scout, out, dp);
}
